// Qwen3Attention_4750233830010
// MI455X (gfx1250) — compile-verified, matrix operand order checked by reading, not run on hardware
//
#include <hip/hip_runtime.h>
#include <hip/hip_bf16.h>

// ---------------- problem constants ----------------
#define HIDDEN   2048
#define SEQ      2048
#define BATCH    2
#define NHEADS   16
#define NKV      8
#define HDIM     128
#define QSIZE    (NHEADS * HDIM)          // 2048
#define KVSIZE   (NKV * HDIM)             // 1024
#define QKV_OUT  (QSIZE + 2 * KVSIZE)     // 4096
#define NTOK     (BATCH * SEQ)            // 4096
#define EPSF     1e-6f
#define LN_THETA 9.210340371976184f       // ln(10000)
#define SCALE    0.08838834764831845f     // 1/sqrt(128)

typedef __attribute__((ext_vector_type(16))) __bf16 v16bf;
typedef __attribute__((ext_vector_type(8)))  __bf16 v8bf;
typedef __attribute__((ext_vector_type(8)))  float  v8f;

__device__ __forceinline__ v8f wmma_bf16(v16bf a, v16bf b, v8f c) {
  // D(f32 16x16) = A(bf16 16x32) * B(bf16 32x16) + C
  return __builtin_amdgcn_wmma_f32_16x16x32_bf16(
      /*neg_a=*/false, a, /*neg_b=*/false, b,
      /*c_mod=*/(short)0, c, /*reuse_a=*/false, /*reuse_b=*/false);
}

// ---------------- async global->LDS copy (CDNA5 ASYNCcnt path) ----------------
// Inline asm per CDNA5 ISA (GLOBAL_LOAD_ASYNC_TO_LDS_B128,
// opcode 98): VDST = per-lane LDS byte address, VADDR = 64-bit global address.
// A generic __shared__ pointer holds the LDS byte offset in its low 32 bits
// (ISA §10.2), so truncation is the conversion.
__device__ __forceinline__ void cp16_to_lds(const __bf16* g, __bf16* l) {
#if defined(__gfx1250__)
  const unsigned lds_addr = (unsigned)(unsigned long long)l;
  asm volatile("global_load_async_to_lds_b128 %0, %1, off"
               :
               : "v"(lds_addr), "v"(g)
               : "memory");
#else
  *(v8bf*)l = *(const v8bf*)g;   // host pass / non-gfx1250 fallback
#endif
}

__device__ __forceinline__ void async_wait_all() {
#if defined(__gfx1250__)
#if __has_builtin(__builtin_amdgcn_s_wait_asynccnt)
  __builtin_amdgcn_s_wait_asynccnt(0);
#else
  asm volatile("s_wait_asynccnt 0x0" ::: "memory");
#endif
#endif
}

// ---------------- fp32 -> bf16 convert ----------------
__global__ void f32_to_bf16_kernel(const float* __restrict__ src,
                                   __bf16* __restrict__ dst, long long n) {
  long long i = (long long)blockIdx.x * blockDim.x + threadIdx.x;
  long long stride = (long long)gridDim.x * blockDim.x;
  for (; i < n; i += stride) dst[i] = (__bf16)src[i];
}

// ---------------- generic bf16 WMMA GEMM, LDS double-buffered ----------------
// C[M,N](f32) = A[M,K](bf16 row-major) x Bw[N,K](bf16 row-major, i.e. B^T)
// block = 256 threads (8 waves); block tile 128x128; wave tile 16x128; K-step 32.
// A/B tiles staged in LDS via async copies; next tile DMA overlaps current WMMAs.
__device__ __forceinline__ void stage_tile_128x32(const __bf16* __restrict__ G,
                                                  int ldg, __bf16 (*L)[32],
                                                  int rowbase, int kk, int tid) {
  // 128x32 bf16 = 8 KB = 512 chunks of 16B; 256 threads x 2 chunks.
#pragma unroll
  for (int i = 0; i < 2; ++i) {
    const int c    = tid + i * 256;      // 0..511
    const int row  = c >> 2;             // 0..127
    const int coff = (c & 3) * 8;        // 0,8,16,24 (bf16 elems)
    cp16_to_lds(G + (size_t)(rowbase + row) * ldg + kk + coff, &L[row][coff]);
  }
}

__global__ void gemm_bf16_wmma(const __bf16* __restrict__ A,
                               const __bf16* __restrict__ Bw,
                               float* __restrict__ C,
                               int M, int N, int K) {
  const int tid  = threadIdx.x;
  const int w    = tid >> 5;
  const int lane = tid & 31;
  const int half = lane >> 4;
  const int l16  = lane & 15;
  const int mblk = blockIdx.y * 128;
  const int n0   = blockIdx.x * 128;

  __shared__ __align__(128) __bf16 Ab[2][128][32];   // 2 x 8 KB
  __shared__ __align__(128) __bf16 Bb[2][128][32];   // 2 x 8 KB

  v8f acc[8] = {};

  // prologue: stage first K-tile
  stage_tile_128x32(A,  K, Ab[0], mblk, 0, tid);
  stage_tile_128x32(Bw, K, Bb[0], n0,   0, tid);
  async_wait_all();
  __syncthreads();

  int buf = 0;
  for (int kk = 0; kk < K; kk += 32) {
    // issue DMA for next tile; overlaps with WMMAs below
    if (kk + 32 < K) {
      stage_tile_128x32(A,  K, Ab[buf ^ 1], mblk, kk + 32, tid);
      stage_tile_128x32(Bw, K, Bb[buf ^ 1], n0,   kk + 32, tid);
    }
    // compute from LDS: A fragment shared per wave, B fragments shared per block
    v16bf af = *(const v16bf*)&Ab[buf][w * 16 + l16][half * 16];
#pragma unroll
    for (int nt = 0; nt < 8; ++nt) {
      v16bf bfr = *(const v16bf*)&Bb[buf][nt * 16 + l16][half * 16];
      acc[nt] = wmma_bf16(af, bfr, acc[nt]);
    }
    async_wait_all();
    __syncthreads();
    buf ^= 1;
  }

#pragma unroll
  for (int nt = 0; nt < 8; ++nt)
#pragma unroll
    for (int r = 0; r < 8; ++r)
      C[(size_t)(mblk + w * 16 + r + 8 * half) * N + n0 + nt * 16 + l16] = acc[nt][r];
}

// ---------------- per-head RMSNorm + RoPE + V transpose ----------------
// qkv_f32: [NTOK, 4096] = [q(2048) | k(1024) | v(1024)]
// qout: [B, 16, S, 128] bf16   kout: [B, 8, S, 128] bf16   vt: [B, 8, 128, S] bf16
__global__ void qknorm_rope_kernel(const float* __restrict__ qkv,
                                   const int* __restrict__ positions,
                                   const float* __restrict__ qw,
                                   const float* __restrict__ kw,
                                   __bf16* __restrict__ qout,
                                   __bf16* __restrict__ kout,
                                   __bf16* __restrict__ vtout) {
  const int b = blockIdx.x / SEQ;
  const int s = blockIdx.x % SEQ;
  const int tid = threadIdx.x;               // 128 threads
  __shared__ float buf[HDIM];
  __shared__ float red[4];

  const float* row = qkv + (size_t)(b * SEQ + s) * QKV_OUT;
  const float pos = (float)positions[b * SEQ + s];
  const int d2 = tid & 63;
  const float freq = __expf(-((float)(2 * d2) / (float)HDIM) * LN_THETA);
  const float ang = pos * freq;
  const float cs = __cosf(ang), sn = __sinf(ang);

  for (int hh = 0; hh < NHEADS + NKV; ++hh) {
    const bool isq = hh < NHEADS;
    const int off = isq ? hh * HDIM : QSIZE + (hh - NHEADS) * HDIM;
    const float x = row[off + tid];
    float ss = x * x;
#pragma unroll
    for (int o = 16; o; o >>= 1) ss += __shfl_xor(ss, o, 32);
    if ((tid & 31) == 0) red[tid >> 5] = ss;
    __syncthreads();
    const float tot = red[0] + red[1] + red[2] + red[3];
    const float rms = rsqrtf(tot * (1.0f / (float)HDIM) + EPSF);
    const float wgt = isq ? qw[tid] : kw[tid];
    buf[tid] = x * rms * wgt;
    __syncthreads();
    float outv;
    if (tid < 64) outv = buf[tid] * cs - buf[tid + 64] * sn;
    else          outv = buf[tid] * cs + buf[tid - 64] * sn;
    if (isq)
      qout[((size_t)(b * NHEADS + hh) * SEQ + s) * HDIM + tid] = (__bf16)outv;
    else
      kout[((size_t)(b * NKV + (hh - NHEADS)) * SEQ + s) * HDIM + tid] = (__bf16)outv;
    __syncthreads();
  }
  // V: convert + transpose to [b, kvh, d, s]
  for (int hh = 0; hh < NKV; ++hh) {
    const float v = row[QSIZE + KVSIZE + hh * HDIM + tid];
    vtout[((size_t)(b * NKV + hh) * HDIM + tid) * SEQ + s] = (__bf16)v;
  }
}

// ---------------- causal flash attention (bf16 WMMA, f32 softmax) ----------------
// grid: (S/128, NHEADS, B), block 256 (8 waves); wave owns a 16-row query tile.
// q: [B,16,S,128]  k: [B,8,S,128]  vt: [B,8,128,S]  out(attn): [NTOK, 2048] bf16
__global__ void flash_attn_kernel(const __bf16* __restrict__ q,
                                  const __bf16* __restrict__ k,
                                  const __bf16* __restrict__ vt,
                                  __bf16* __restrict__ attn) {
  const int w    = threadIdx.x >> 5;
  const int lane = threadIdx.x & 31;
  const int half = lane >> 4;
  const int l16  = lane & 15;
  const int b    = blockIdx.z;
  const int h    = blockIdx.y;
  const int kvh  = h >> 1;                   // GQA: 2 q-heads per kv-head
  const int qbase = blockIdx.x * 128 + w * 16;

  __shared__ __align__(64) __bf16 plds[8][16][32]; // 1 KB per wave

  // Q fragments: 16x128 strip, 4 K-chunks of 32
  v16bf qf[4];
  const __bf16* Qp = q + ((size_t)(b * NHEADS + h) * SEQ + qbase + l16) * HDIM;
#pragma unroll
  for (int dc = 0; dc < 4; ++dc)
    qf[dc] = *(const v16bf*)(Qp + dc * 32 + half * 16);

  const __bf16* Kb = k  + (size_t)(b * NKV + kvh) * SEQ * HDIM;
  const __bf16* Vb = vt + (size_t)(b * NKV + kvh) * HDIM * SEQ;

  v8f o[8] = {};
  float mi[8], li[8];
#pragma unroll
  for (int r = 0; r < 8; ++r) { mi[r] = -3.0e38f; li[r] = 0.0f; }

  const int qend = qbase + 16;               // causal bound (per-wave trip count!)
  for (int kb = 0; kb < qend; kb += 32) {
    // scores: 16x32 = two 16x16 f32 accumulators
    v8f s0 = {}, s1 = {};
#pragma unroll
    for (int dc = 0; dc < 4; ++dc) {
      const int koff = dc * 32 + half * 16;
      v16bf kf0 = *(const v16bf*)(Kb + (size_t)(kb + l16) * HDIM + koff);
      v16bf kf1 = *(const v16bf*)(Kb + (size_t)(kb + 16 + l16) * HDIM + koff);
      s0 = wmma_bf16(qf[dc], kf0, s0);
      s1 = wmma_bf16(qf[dc], kf1, s1);
    }
    // online softmax (rows live at r + 8*half, cols = lane within 16-half)
    float p0[8], p1[8], corr[8];
#pragma unroll
    for (int r = 0; r < 8; ++r) {
      const int row = qbase + r + 8 * half;
      float a = s0[r] * SCALE; if (kb + l16 > row)      a = -3.0e38f;
      float c = s1[r] * SCALE; if (kb + 16 + l16 > row) c = -3.0e38f;
      float rm = fmaxf(a, c);
#pragma unroll
      for (int off = 8; off; off >>= 1) rm = fmaxf(rm, __shfl_xor(rm, off, 16));
      const float mnew = fmaxf(mi[r], rm);
      corr[r] = __expf(mi[r] - mnew);
      a = __expf(a - mnew);
      c = __expf(c - mnew);
      float rs = a + c;
#pragma unroll
      for (int off = 8; off; off >>= 1) rs += __shfl_xor(rs, off, 16);
      li[r] = li[r] * corr[r] + rs;
      mi[r] = mnew;
      p0[r] = a; p1[r] = c;
    }
#pragma unroll
    for (int t = 0; t < 8; ++t)
#pragma unroll
      for (int r = 0; r < 8; ++r) o[t][r] *= corr[r];

    // C-layout -> A-layout via per-wave LDS (LDS is in-order within a wave)
#pragma unroll
    for (int r = 0; r < 8; ++r) {
      plds[w][r + 8 * half][l16]      = (__bf16)p0[r];
      plds[w][r + 8 * half][16 + l16] = (__bf16)p1[r];
    }
    v16bf pf = *(const v16bf*)(&plds[w][l16][half * 16]);

    // O += P(16x32) x V^T-fragments(32x16 per d-tile)
#pragma unroll
    for (int dt = 0; dt < 8; ++dt) {
      const __bf16* vp = Vb + (size_t)(dt * 16 + l16) * SEQ + kb + half * 16;
      v16bf vf = *(const v16bf*)vp;
      o[dt] = wmma_bf16(pf, vf, o[dt]);
    }
  }

  // normalize + store token-major bf16 for the output projection
  float inv[8];
#pragma unroll
  for (int r = 0; r < 8; ++r) inv[r] = 1.0f / li[r];
#pragma unroll
  for (int dt = 0; dt < 8; ++dt)
#pragma unroll
    for (int r = 0; r < 8; ++r) {
      const size_t row = (size_t)(b * SEQ + qbase + r + 8 * half);
      attn[row * QSIZE + h * HDIM + dt * 16 + l16] = (__bf16)(o[dt][r] * inv[r]);
    }
}

// ---------------- host launcher ----------------
extern "C" void kernel_launch(void* const* d_in, const int* in_sizes, int n_in,
                              void* d_out, int out_size, void* d_ws, size_t ws_size,
                              hipStream_t stream) {
  const int*   positions = (const int*)d_in[0];
  const float* hidden    = (const float*)d_in[1];   // [B,S,2048]
  const float* w_qkv     = (const float*)d_in[2];   // [4096,2048]
  const float* w_o       = (const float*)d_in[3];   // [2048,2048]
  const float* q_norm_w  = (const float*)d_in[4];
  const float* k_norm_w  = (const float*)d_in[5];
  float* out = (float*)d_out;                       // [B,S,2048]

  char* ws = (char*)d_ws;
  // workspace layout (bytes)
  constexpr size_t SZ_HID_BF  = (size_t)NTOK * HIDDEN * 2;        // 16 MB
  constexpr size_t SZ_WQKV_BF = (size_t)QKV_OUT * HIDDEN * 2;     // 16 MB
  constexpr size_t SZ_WO_BF   = (size_t)HIDDEN * QSIZE * 2;       //  8 MB
  constexpr size_t SZ_QKV_F32 = (size_t)NTOK * QKV_OUT * 4;       // 64 MB
  constexpr size_t SZ_Q_BF    = (size_t)BATCH * NHEADS * SEQ * HDIM * 2; // 16 MB
  constexpr size_t SZ_K_BF    = (size_t)BATCH * NKV * SEQ * HDIM * 2;    //  8 MB
  constexpr size_t SZ_VT_BF   = SZ_K_BF;                          //  8 MB
  __bf16* hid_bf  = (__bf16*)(ws);
  __bf16* wqkv_bf = (__bf16*)(ws + SZ_HID_BF);
  __bf16* wo_bf   = (__bf16*)(ws + SZ_HID_BF + SZ_WQKV_BF);
  float*  qkv_f32 = (float*) (ws + SZ_HID_BF + SZ_WQKV_BF + SZ_WO_BF);
  __bf16* q_bf    = (__bf16*)(ws + SZ_HID_BF + SZ_WQKV_BF + SZ_WO_BF + SZ_QKV_F32);
  __bf16* k_bf    = (__bf16*)((char*)q_bf + SZ_Q_BF);
  __bf16* vt_bf   = (__bf16*)((char*)k_bf + SZ_K_BF);
  __bf16* attn_bf = (__bf16*)((char*)vt_bf + SZ_VT_BF);

  // 1) convert operands to bf16
  f32_to_bf16_kernel<<<4096, 256, 0, stream>>>(hidden, hid_bf, (long long)NTOK * HIDDEN);
  f32_to_bf16_kernel<<<4096, 256, 0, stream>>>(w_qkv, wqkv_bf, (long long)QKV_OUT * HIDDEN);
  f32_to_bf16_kernel<<<4096, 256, 0, stream>>>(w_o, wo_bf, (long long)HIDDEN * QSIZE);

  // 2) QKV projection: [4096,4096] = hid[4096,2048] x w_qkv^T
  gemm_bf16_wmma<<<dim3(QKV_OUT / 128, NTOK / 128), 256, 0, stream>>>(
      hid_bf, wqkv_bf, qkv_f32, NTOK, QKV_OUT, HIDDEN);

  // 3) qk RMSNorm + RoPE + V transpose
  qknorm_rope_kernel<<<NTOK, 128, 0, stream>>>(
      qkv_f32, positions, q_norm_w, k_norm_w, q_bf, k_bf, vt_bf);

  // 4) causal flash attention
  flash_attn_kernel<<<dim3(SEQ / 128, NHEADS, BATCH), 256, 0, stream>>>(
      q_bf, k_bf, vt_bf, attn_bf);

  // 5) output projection: out[4096,2048] = attn[4096,2048] x w_o^T
  gemm_bf16_wmma<<<dim3(HIDDEN / 128, NTOK / 128), 256, 0, stream>>>(
      attn_bf, wo_bf, out, NTOK, HIDDEN, QSIZE);
}
